// BiLSTM_CRF_66812511257007
// MI455X (gfx1250) — matrix operand order repaired and checked by reading, not compiled, not run on hardware
//
#include <hip/hip_runtime.h>
#include <hip/hip_bf16.h>
#include <math.h>

typedef __attribute__((ext_vector_type(16))) _Float16 v16h;
typedef __attribute__((ext_vector_type(8)))  _Float16 v8h;
typedef __attribute__((ext_vector_type(8)))  float    v8f;

#define T_LEN 2048
#define E_DIM 100
#define KPAD  128
#define H_DIM 128
#define G3    384          // 3*H
#define NT    24           // 384/16 N-tiles
#define S_DIM 6
#define B_LAST 63

// ---------------------------------------------------------------------------
// K1a: gather embedding row for batch 63, convert to f16, pad K 100 -> 128
// ---------------------------------------------------------------------------
__global__ void k_gather_x(const int* __restrict__ sent,
                           const float* __restrict__ emb,
                           _Float16* __restrict__ xb16) {
    int i = blockIdx.x * blockDim.x + threadIdx.x;   // < T_LEN*KPAD
    int t = i >> 7, k = i & 127;
    int v = sent[B_LAST * T_LEN + t];
    float val = (k < E_DIM) ? emb[v * E_DIM + k] : 0.0f;
    xb16[i] = (_Float16)val;
}

// ---------------------------------------------------------------------------
// K1b/K1c: pack W (row-major [384, srcK]) as B-matrix (W.T, K padded to 128)
// into per-lane-contiguous WMMA B-fragment order:
//   lane = (k%32 / 16)*16 + n%16,  elem = k%16
// ---------------------------------------------------------------------------
__global__ void k_pack_b(const float* __restrict__ Wf,
                         const float* __restrict__ Wb,
                         _Float16* __restrict__ pack, int srcK) {
    int i = blockIdx.x * blockDim.x + threadIdx.x;   // < 2*G3*KPAD
    int dir = i / (G3 * KPAD);
    int r   = i % (G3 * KPAD);
    int n = r / KPAD, k = r % KPAD;
    const float* W = dir ? Wb : Wf;
    float val = (k < srcK) ? W[n * srcK + k] : 0.0f;
    int nt = n >> 4, nl = n & 15;
    int ks = k >> 5, kw = k & 31;
    int lane = ((kw >> 4) << 4) + nl;
    int e    = kw & 15;
    pack[(((dir * NT + nt) * 4 + ks) << 9) + lane * 16 + e] = (_Float16)val;
}

// ---------------------------------------------------------------------------
// K2: gi[dir][t][n] = x[t,:] @ Wih.T + bih   via WMMA f32_16x16x32_f16
// grid (128 M-tiles, 6, 2 dirs), block 128 (4 waves, one N-tile each)
// ---------------------------------------------------------------------------
__global__ void __launch_bounds__(128)
k_gemm_gi(const _Float16* __restrict__ xb16,
          const _Float16* __restrict__ bpack,
          const float* __restrict__ bf_ih, const float* __restrict__ bb_ih,
          float* __restrict__ gi) {
    int lane = threadIdx.x & 31;
    int w    = threadIdx.x >> 5;
    int nt   = blockIdx.y * 4 + w;
    int mt   = blockIdx.x;
    int dir  = blockIdx.z;
    int nl   = lane & 15;
    int m    = mt * 16 + nl;
    int kbase = (lane < 16) ? 0 : 16;
    int n = nt * 16 + nl;

    const float* bih = dir ? bb_ih : bf_ih;
    float bv = bih[n];
    v8f acc;
#pragma unroll
    for (int g = 0; g < 8; ++g) acc[g] = bv;     // fold bias into C

    const _Float16* arow  = xb16 + m * KPAD;
    const _Float16* bbase = bpack + (((dir * NT + nt) * 4) << 9) + lane * 16;

#pragma unroll
    for (int ks = 0; ks < 4; ++ks) {
        v8h lo = *(const v8h*)(arow + ks * 32 + kbase);
        v8h hi = *(const v8h*)(arow + ks * 32 + kbase + 8);
        v16h a;
#pragma unroll
        for (int e = 0; e < 8; ++e) { a[e] = lo[e]; a[8 + e] = hi[e]; }
        v16h b = *(const v16h*)(bbase + (ks << 9));
        acc = __builtin_amdgcn_wmma_f32_16x16x32_f16(false, a, false, b,
                                                     (short)0, acc, false, false);
    }

    float* gd = gi + (size_t)dir * T_LEN * G3;
    int mrow = mt * 16 + ((lane >= 16) ? 8 : 0);
#pragma unroll
    for (int g = 0; g < 8; ++g)
        gd[(size_t)(mrow + g) * G3 + n] = acc[g];
}

// ---------------------------------------------------------------------------
// K3: sequential GRU recurrence for batch 63, one block per direction.
// gh = h @ Whh.T via WMMA with h replicated into all 16 A rows
// (every lane loads the row-0 K pattern for its half-wave -> A[m][k]=h[k]).
// 384 threads = 12 waves x 2 N-tiles; B fragments live in registers for
// all 2048 steps. Gates on threads 0..127; h round-trips through LDS f16.
// gi rows are software-prefetched 2 steps ahead (global_prefetch_b8).
// ---------------------------------------------------------------------------
__global__ void __launch_bounds__(384)
k_recur(const _Float16* __restrict__ whhpack, const float* __restrict__ gi,
        const float* __restrict__ h0,
        const float* __restrict__ bf_hh, const float* __restrict__ bb_hh,
        float* __restrict__ hbuf) {
    __shared__ _Float16 h16[KPAD];
    __shared__ float    gh_s[G3];

    int dir  = blockIdx.x;
    int tid  = threadIdx.x;
    int lane = tid & 31;
    int w    = tid >> 5;          // 0..11
    int nl   = lane & 15;
    int nt0  = w * 2, nt1 = w * 2 + 1;

    // Preload B fragments for Whh.T (2 N-tiles x 4 K-slices) into registers.
    v16h bfrag[8];
    {
        const _Float16* base = whhpack + (((size_t)dir * NT) << 11);
#pragma unroll
        for (int i = 0; i < 2; ++i)
#pragma unroll
            for (int ks = 0; ks < 4; ++ks)
                bfrag[i * 4 + ks] =
                    *(const v16h*)(base + ((((w * 2 + i) * 4) + ks) << 9) + lane * 16);
    }

    const float* bhh = dir ? bb_hh : bf_hh;
    const float* gd  = gi + (size_t)dir * T_LEN * G3;
    float* hd        = hbuf + (size_t)dir * T_LEN * H_DIM;

    float hreg = 0.f, b_r = 0.f, b_z = 0.f, b_n = 0.f;
    if (tid < H_DIM) {
        hreg = h0[dir * 64 * H_DIM + B_LAST * H_DIM + tid];
        b_r = bhh[tid];
        b_z = bhh[H_DIM + tid];
        b_n = bhh[2 * H_DIM + tid];
        h16[tid] = (_Float16)hreg;
    }
    __syncthreads();

    int kbase = (lane < 16) ? 0 : 16;

    for (int s = 0; s < T_LEN; ++s) {
        int t = dir ? (T_LEN - 1 - s) : s;

        // Prefetch the gi row two steps ahead (1536 B = 12 cachelines);
        // threads 0..11 each touch one cacheline.
        if (s + 2 < T_LEN && tid < 12) {
            int t2 = dir ? (T_LEN - 1 - (s + 2)) : (s + 2);
            __builtin_prefetch(gd + (size_t)t2 * G3 + tid * 32, 0, 1);
        }

        // Issue gi-row loads early (independent of recurrence).
        float gi_r = 0.f, gi_z = 0.f, gi_n = 0.f;
        if (tid < H_DIM) {
            const float* gr = gd + (size_t)t * G3;
            gi_r = gr[tid];
            gi_z = gr[H_DIM + tid];
            gi_n = gr[2 * H_DIM + tid];
        }

        v8f acc0 = {}; v8f acc1 = {};
#pragma unroll
        for (int ks = 0; ks < 4; ++ks) {
            v8h lo = *(const v8h*)&h16[ks * 32 + kbase];
            v8h hi = *(const v8h*)&h16[ks * 32 + kbase + 8];
            v16h a;
#pragma unroll
            for (int e = 0; e < 8; ++e) { a[e] = lo[e]; a[8 + e] = hi[e]; }
            acc0 = __builtin_amdgcn_wmma_f32_16x16x32_f16(false, a, false, bfrag[ks],
                                                          (short)0, acc0, false, false);
            acc1 = __builtin_amdgcn_wmma_f32_16x16x32_f16(false, a, false, bfrag[4 + ks],
                                                          (short)0, acc1, false, false);
        }

        // D row 0 lives in acc VGPR0, lanes 0..15 (N = lane).
        if (lane < 16) {
            gh_s[nt0 * 16 + nl] = acc0[0];
            gh_s[nt1 * 16 + nl] = acc1[0];
        }
        __syncthreads();

        if (tid < H_DIM) {
            float r = 1.0f / (1.0f + __expf(-(gi_r + gh_s[tid] + b_r)));
            float z = 1.0f / (1.0f + __expf(-(gi_z + gh_s[H_DIM + tid] + b_z)));
            float y = gi_n + r * (gh_s[2 * H_DIM + tid] + b_n);
            float e2 = __expf(-2.0f * y);                 // tanh via single v_exp_f32
            float n = (1.0f - e2) / (1.0f + e2);
            hreg = (1.0f - z) * n + z * hreg;
            hd[(size_t)t * H_DIM + tid] = hreg;
            h16[tid] = (_Float16)hreg;
        }
        __syncthreads();
    }
}

// ---------------------------------------------------------------------------
// K4: logits + softmax for batch 63. One thread per timestep; W_out in LDS.
// ---------------------------------------------------------------------------
__global__ void __launch_bounds__(256)
k_logits(const float* __restrict__ hbuf, const float* __restrict__ Wout,
         const float* __restrict__ bout, float* __restrict__ probs) {
    __shared__ float Ws[S_DIM * 2 * H_DIM];
    __shared__ float bs[S_DIM];
    for (int i = threadIdx.x; i < S_DIM * 2 * H_DIM; i += blockDim.x) Ws[i] = Wout[i];
    if (threadIdx.x < S_DIM) bs[threadIdx.x] = bout[threadIdx.x];
    __syncthreads();

    int t = blockIdx.x * blockDim.x + threadIdx.x;
    const float* hf = hbuf + (size_t)t * H_DIM;
    const float* hb = hbuf + (size_t)T_LEN * H_DIM + (size_t)t * H_DIM;

    float acc[S_DIM];
#pragma unroll
    for (int s2 = 0; s2 < S_DIM; ++s2) acc[s2] = bs[s2];
    for (int c = 0; c < H_DIM; ++c) {
        float vF = hf[c], vB = hb[c];
#pragma unroll
        for (int s2 = 0; s2 < S_DIM; ++s2)
            acc[s2] += vF * Ws[s2 * 2 * H_DIM + c] + vB * Ws[s2 * 2 * H_DIM + H_DIM + c];
    }
    float mx = acc[0];
#pragma unroll
    for (int s2 = 1; s2 < S_DIM; ++s2) mx = fmaxf(mx, acc[s2]);
    float sum = 0.f, ex[S_DIM];
#pragma unroll
    for (int s2 = 0; s2 < S_DIM; ++s2) { ex[s2] = __expf(acc[s2] - mx); sum += ex[s2]; }
    float inv = 1.0f / sum;
#pragma unroll
    for (int s2 = 0; s2 < S_DIM; ++s2) probs[t * S_DIM + s2] = ex[s2] * inv;
}

// ---------------------------------------------------------------------------
// K5: Viterbi for batch 63. Single wave; backpointers kept entirely in LDS.
// ---------------------------------------------------------------------------
__global__ void k_viterbi(const float* __restrict__ probs,
                          const float* __restrict__ trans,
                          float* __restrict__ out) {
    __shared__ float trellis[S_DIM];
    __shared__ float tr_s[S_DIM * S_DIM];
    __shared__ short bp[T_LEN * S_DIM];     // 24 KB
    int j = threadIdx.x;
    if (j < S_DIM * S_DIM) tr_s[j] = trans[j];
    if (j < S_DIM) trellis[j] = probs[j];
    __syncthreads();

    for (int t = 1; t < T_LEN; ++t) {
        float best = -1e30f; int bi = 0;
        if (j < S_DIM) {
#pragma unroll
            for (int i = 0; i < S_DIM; ++i) {
                float v = trellis[i] + tr_s[i * S_DIM + j];
                if (v > best) { best = v; bi = i; }
            }
        }
        __syncthreads();
        if (j < S_DIM) {
            trellis[j] = probs[t * S_DIM + j] + best;
            bp[t * S_DIM + j] = (short)bi;
        }
        __syncthreads();
    }

    if (j == 0) {
        float sc = trellis[0]; int last = 0;
#pragma unroll
        for (int i = 1; i < S_DIM; ++i)
            if (trellis[i] > sc) { sc = trellis[i]; last = i; }
        out[0] = sc;
        out[1 + (T_LEN - 1)] = (float)last;
        int cur = last;
        for (int t = T_LEN - 1; t >= 1; --t) {
            cur = (int)bp[t * S_DIM + cur];
            out[1 + (t - 1)] = (float)cur;
        }
    }
}

// ---------------------------------------------------------------------------
extern "C" void kernel_launch(void* const* d_in, const int* in_sizes, int n_in,
                              void* d_out, int out_size, void* d_ws, size_t ws_size,
                              hipStream_t stream) {
    (void)in_sizes; (void)n_in; (void)out_size; (void)ws_size;
    const int*   sent  = (const int*)d_in[0];
    const float* emb   = (const float*)d_in[1];
    const float* h0    = (const float*)d_in[2];
    const float* Wf_ih = (const float*)d_in[3];
    const float* Wf_hh = (const float*)d_in[4];
    const float* bf_ih = (const float*)d_in[5];
    const float* bf_hh = (const float*)d_in[6];
    const float* Wb_ih = (const float*)d_in[7];
    const float* Wb_hh = (const float*)d_in[8];
    const float* bb_ih = (const float*)d_in[9];
    const float* bb_hh = (const float*)d_in[10];
    const float* W_out = (const float*)d_in[11];
    const float* b_out = (const float*)d_in[12];
    const float* trans = (const float*)d_in[13];

    char* ws = (char*)d_ws;
    _Float16* xb16    = (_Float16*)(ws + 0);        // 2048*128*2           = 524288
    _Float16* giBpack = (_Float16*)(ws + 524288);   // 2*24*4*512*2         = 196608
    _Float16* whhPack = (_Float16*)(ws + 720896);   // 2*24*4*512*2         = 196608
    float*    gi      = (float*)   (ws + 917504);   // 2*2048*384*4         = 6291456
    float*    hbuf    = (float*)   (ws + 7208960);  // 2*2048*128*4         = 2097152
    float*    probs   = (float*)   (ws + 9306112);  // 2048*6*4             = 49152

    k_gather_x<<<(T_LEN * KPAD) / 256, 256, 0, stream>>>(sent, emb, xb16);
    k_pack_b<<<(2 * G3 * KPAD) / 256, 256, 0, stream>>>(Wf_ih, Wb_ih, giBpack, E_DIM);
    k_pack_b<<<(2 * G3 * KPAD) / 256, 256, 0, stream>>>(Wf_hh, Wb_hh, whhPack, H_DIM);
    k_gemm_gi<<<dim3(T_LEN / 16, NT / 4, 2), 128, 0, stream>>>(xb16, giBpack, bf_ih, bb_ih, gi);
    k_recur<<<2, 384, 0, stream>>>(whhPack, gi, h0, bf_hh, bb_hh, hbuf);
    k_logits<<<T_LEN / 256, 256, 0, stream>>>(hbuf, W_out, b_out, probs);
    k_viterbi<<<1, 32, 0, stream>>>(probs, trans, (float*)d_out);
}
